// MultipleShooting_52140902974144
// MI455X (gfx1250) — matrix operand order repaired and checked by reading, not compiled, not run on hardware
//
#include <hip/hip_runtime.h>
#include <hip/hip_bf16.h>

typedef __attribute__((ext_vector_type(16))) __bf16 v16bf;
typedef __attribute__((ext_vector_type(8)))  float  v8f;

#define DD   128      // state dimension
#define HH   256      // hidden width
#define LTOT 262144   // time-grid length
#define SUB  1024     // steps per shooting segment
#define MT   16       // segments per block (WMMA M tile)

union AFrag { uint4 q[2]; v16bf v; };

// ---- A fragment from bf16 row-major LDS buffer (row stride `ld` elements) ----
// ISA 16-bit A 16x32 layout: lane half g needs K = kb*32 + 8g + [0..8) in the
// low 8 bf16 and K = kb*32 + 16 + 8g + [0..8) in the high 8 -> two contiguous
// 16-byte chunks, i.e. two ds_load_b128 with zero conversion work.
__device__ __forceinline__ v16bf load_a(const __bf16* X, int ld, int m, int g, int kb) {
    AFrag a;
    a.q[0] = *(const uint4*)(X + m * ld + kb * 32 + 8 * g);
    a.q[1] = *(const uint4*)(X + m * ld + kb * 32 + 16 + 8 * g);
    return a.v;
}

// ---- B fragment from global f32 weights W[K, N] (row stride ldn) ----
// 16-bit B 32x16 layout matching A: lane half g covers K = 8g+[0..8) and 16+8g+[0..8);
// VGPR v holds K = 8g + 2v (+8 if v>=4), +1; column N = lane&15.
__device__ __forceinline__ v16bf load_b(const float* W, int ldn, int kb, int g, int n) {
    v16bf b;
#pragma unroll
    for (int v = 0; v < 8; ++v) {
        int k = kb * 32 + 2 * v + 8 * g + ((v >= 4) ? 8 : 0);
        b[2 * v]     = (__bf16)W[k * ldn + n];
        b[2 * v + 1] = (__bf16)W[(k + 1) * ldn + n];
    }
    return b;
}

__device__ __forceinline__ float fast_tanh(float x) {
#if __has_builtin(__builtin_amdgcn_tanhf)
    return __builtin_amdgcn_tanhf(x);      // gfx1250 v_tanh_f32
#else
    float e = __expf(2.0f * x);
    return 1.0f - 2.0f / (e + 1.0f);
#endif
}

__global__ __launch_bounds__(256, 1)
void ms_rk4_wmma_kernel(const float* __restrict__ t,  const float* __restrict__ y,
                        const float* __restrict__ W1, const float* __restrict__ b1,
                        const float* __restrict__ W2, const float* __restrict__ b2,
                        float* __restrict__ out)
{
    __shared__ float  ys[MT * DD];   // current state (fp32)
    __shared__ float  kk[MT * DD];   // RHS output (k_i, fp32)
    __shared__ float  ac[MT * DD];   // RK4 k-accumulator (fp32)
    __shared__ __bf16 xb[MT * DD];   // RHS input staged as bf16 (WMMA-1 A source)
    __shared__ __bf16 hb[MT * HH];   // tanh(hidden) in bf16 (WMMA-2 A source)

    const int tid  = threadIdx.x;
    const int lane = tid & 31;
    const int wv   = tid >> 5;        // wave 0..7
    const int nl   = lane & 15;       // N column / A row within half
    const int g    = lane >> 4;       // lane half
    const int blk  = blockIdx.x;      // 0..15, segment tile

    const float dt  = t[1] - t[0];
    const float dt6 = dt * (1.0f / 6.0f);

    // ---- loop-invariant weight B-fragments, held in VGPRs for whole chain ----
    v16bf B1f[2][4];                  // wave's hidden cols [32w,32w+32), K=128
#pragma unroll
    for (int tI = 0; tI < 2; ++tI) {
        int n1 = 16 * (2 * wv + tI) + nl;
#pragma unroll
        for (int kb = 0; kb < 4; ++kb) B1f[tI][kb] = load_b(W1, HH, kb, g, n1);
    }
    v16bf B2f[8];                     // wave's out cols [16w,16w+16), K=256
    const int n2 = 16 * wv + nl;
#pragma unroll
    for (int kb = 0; kb < 8; ++kb) B2f[kb] = load_b(W2, DD, kb, g, n2);

    const float b1v0 = b1[16 * (2 * wv + 0) + nl];
    const float b1v1 = b1[16 * (2 * wv + 1) + nl];
    const float b2v  = b2[n2];

    // ---- init shooting vars from signal at linspace'd indices ----
    {
        __bf16 xtmp[8];
#pragma unroll
        for (int e = 0; e < 8; ++e) {
            int i = tid * 8 + e;
            int m = i >> 7, d = i & (DD - 1);
            int seg = blk * MT + m;
            int col = (int)(((double)seg * (double)(LTOT - 1)) / 256.0);
            float v = y[(long long)d * LTOT + col];
            ys[i] = v;
            xtmp[e] = (__bf16)v;
            out[(long long)d * LTOT + (long long)seg * SUB + 0] = v;      // pred[0]
            if (seg >= 1)                                                 // shooting_vars[1:]
                out[(long long)DD * LTOT + 255 * DD + (seg - 1) * DD + d] = v;
        }
        *(uint4*)(xb + tid * 8) = *(uint4*)xtmp;
    }
    __syncthreads();

    // ---- RHS: kk = tanh(xb @ W1 + b1) @ W2 + b2  (xb = bf16 staged input) ----
    auto rhs = [&]() {
#pragma unroll
        for (int tI = 0; tI < 2; ++tI) {
            v8f acc = {};
#pragma unroll
            for (int kb = 0; kb < 4; ++kb) {
                v16bf a = load_a(xb, DD, nl, g, kb);
                acc = __builtin_amdgcn_wmma_f32_16x16x32_bf16(
                          false, a, false, B1f[tI][kb], (short)0, acc, false, false);
            }
            float bb = (tI == 0) ? b1v0 : b1v1;
            int   nh = 16 * (2 * wv + tI) + nl;
#pragma unroll
            for (int r = 0; r < 8; ++r) {
                int m = r + 8 * g;                    // C/D row = r + 8*(lane>>4)
                hb[m * HH + nh] = (__bf16)fast_tanh(acc[r] + bb);
            }
        }
        __syncthreads();

        v8f acc2 = {};
#pragma unroll
        for (int kb = 0; kb < 8; ++kb) {
            v16bf a = load_a(hb, HH, nl, g, kb);
            acc2 = __builtin_amdgcn_wmma_f32_16x16x32_bf16(
                       false, a, false, B2f[kb], (short)0, acc2, false, false);
        }
#pragma unroll
        for (int r = 0; r < 8; ++r) {
            int m = r + 8 * g;
            kk[m * DD + n2] = acc2[r] + b2v;
        }
        __syncthreads();
    };

    // elementwise helper: ac (+)= w*k ; xb = bf16(ys + c*k)
    auto mix = [&](float wacc, float c, bool first) {
        __bf16 xtmp[8];
#pragma unroll
        for (int e = 0; e < 8; ++e) {
            int i = tid * 8 + e;
            float k = kk[i];
            if (first) ac[i] = k; else ac[i] += wacc * k;
            xtmp[e] = (__bf16)(ys[i] + c * k);
        }
        *(uint4*)(xb + tid * 8) = *(uint4*)xtmp;
        __syncthreads();
    };

    // ---- 1023 sequential RK4 steps ----
    for (int s = 1; s < SUB; ++s) {
        rhs();                                   // k1 (reads xb == bf16(ys))
        mix(1.0f, 0.5f * dt, true);
        rhs();                                   // k2
        mix(2.0f, 0.5f * dt, false);
        rhs();                                   // k3
        mix(2.0f, dt, false);
        rhs();                                   // k4 -> update state + store traj
        {
            __bf16 xtmp[8];
#pragma unroll
            for (int e = 0; e < 8; ++e) {
                int i  = tid * 8 + e;
                float yn = ys[i] + dt6 * (ac[i] + kk[i]);
                ys[i] = yn;
                xtmp[e] = (__bf16)yn;
                int m = i >> 7, d = i & (DD - 1);
                int seg = blk * MT + m;
                out[(long long)d * LTOT + (long long)seg * SUB + s] = yn;
            }
            *(uint4*)(xb + tid * 8) = *(uint4*)xtmp;
        }
        __syncthreads();
    }

    // ---- shooting_end_values[:-1] ----
#pragma unroll
    for (int e = 0; e < 8; ++e) {
        int i = tid * 8 + e;
        int m = i >> 7, d = i & (DD - 1);
        int seg = blk * MT + m;
        if (seg < 255)
            out[(long long)DD * LTOT + seg * DD + d] = ys[i];
    }
}

extern "C" void kernel_launch(void* const* d_in, const int* in_sizes, int n_in,
                              void* d_out, int out_size, void* d_ws, size_t ws_size,
                              hipStream_t stream) {
    const float* t  = (const float*)d_in[0];
    const float* y  = (const float*)d_in[1];
    const float* W1 = (const float*)d_in[2];
    const float* b1 = (const float*)d_in[3];
    const float* W2 = (const float*)d_in[4];
    const float* b2 = (const float*)d_in[5];
    float* out = (float*)d_out;
    // 256 segments / 16 per block = 16 blocks; 256 threads = 8 wave32 per block
    ms_rk4_wmma_kernel<<<dim3(16), dim3(256), 0, stream>>>(t, y, W1, b1, W2, b2, out);
}
